// DynamicsEnsemble_46076409151814
// MI455X (gfx1250) — matrix operand order repaired and checked by reading, not compiled, not run on hardware
//
#include <hip/hip_runtime.h>
#include <hip/hip_bf16.h>

// ---------- types ----------
typedef __bf16 v16bf __attribute__((ext_vector_type(16)));
typedef float  v8f   __attribute__((ext_vector_type(8)));
typedef unsigned int u32x4 __attribute__((ext_vector_type(4)));

// ---------- problem constants ----------
#define B_TOTAL 32768
#define SDIM 64
#define ADIM 16
#define IN_PAD 96        // 80 padded to 3 k-tiles of 32
#define IN_LD 104        // padded LDS row stride (bank-conflict-free, 16B aligned)
#define HID 512
#define HID_LD 520       // padded LDS row stride
#define ENS 7
#define ODIM 130
#define OPAD 144         // 130 padded to 9 n-tiles of 16
#define TILE_M 64
#define TILES (B_TOTAL / TILE_M)   // 512

// ---------- workspace layout (bytes) ----------
#define OFF_CNT  0
#define OFF_ROWS 1024
#define N_W1F (ENS * 32 * 3 * 256)    // dwords of frag-swizzled weights
#define N_W2F (ENS * 32 * 16 * 256)
#define N_W3F (ENS * 9 * 16 * 256)
#define OFF_W1F (OFF_ROWS + ENS * B_TOTAL * 4)
#define OFF_W2F (OFF_W1F + N_W1F * 4)
#define OFF_W3F (OFF_W2F + N_W2F * 4)

#define SMEM_BYTES (2 * TILE_M * HID_LD * 2)   // H1 + X/H2, bf16: 133120 B

__device__ __forceinline__ unsigned short f2bf(float f) {  // host-side-free RNE pack
  unsigned u = __float_as_uint(f);
  u += 0x7FFFu + ((u >> 16) & 1u);
  return (unsigned short)(u >> 16);
}

// ---------- pass 0: zero counters ----------
__global__ void k_zero(int* cnt) {
  if (threadIdx.x < 8) cnt[threadIdx.x] = 0;
}

// ---------- pass 1: bucket rows by ensemble index ----------
__global__ void k_scatter(const int* __restrict__ idx, int* __restrict__ cnt,
                          int* __restrict__ rows) {
  int b = blockIdx.x * 256 + threadIdx.x;
  if (b >= B_TOTAL) return;
  int e = idx[b];
  if (e >= 0 && e < ENS) {
    int pos = atomicAdd(&cnt[e], 1);
    rows[e * B_TOTAL + pos] = b;
  }
}

// ---------- pass 2: fp32 -> bf16 weights, pre-swizzled into WMMA B-fragment order.
// B frag (16-bit B 32x16, same k order as A): lane L -> N=L&15, hi=L>>4;
// dwords 0..3: K=hi*8+2j,+1; dwords 4..7: K=16+hi*8+2(j-4),+1.  dst[((e*NT+nt)*KT+kt)*256 + lane*8 + j]
__global__ void k_convert(const float* __restrict__ W, unsigned* __restrict__ dst,
                          int Ksrc, int Nsrc, int NT, int KT, int total) {
  int d = blockIdx.x * 256 + threadIdx.x;
  if (d >= total) return;
  int j = d & 7, lane = (d >> 3) & 31, fr = d >> 8;
  int kt = fr % KT, q = fr / KT;
  int nt = q % NT, e = q / NT;
  int n = nt * 16 + (lane & 15);
  int k = kt * 32 + ((lane >> 4) << 3) + ((j >> 2) << 3) + (j << 1);
  float lo = 0.f, hi = 0.f;
  if (n < Nsrc) {
    if (k < Ksrc)     lo = W[(e * Ksrc + k) * Nsrc + n];
    if (k + 1 < Ksrc) hi = W[(e * Ksrc + k + 1) * Nsrc + n];
  }
  dst[d] = (unsigned)f2bf(lo) | ((unsigned)f2bf(hi) << 16);
}

// ---------- fragment loaders ----------
// A frag (16-bit A 16x32): lane m=L&15, hi=L>>4; dwords 0..3: k=hi*8+{0..7};
// dwords 4..7: k=16+hi*8+{0..7}  -> two 16B-contiguous ds_load_b128
__device__ __forceinline__ v16bf load_a(const __bf16* stripe, int ldk,
                                        int kt, int lane) {
  int m = lane & 15, hi = (lane >> 4) & 1;
  union { v16bf v; u32x4 u[2]; } f;
  const __bf16* p = stripe + m * ldk + kt * 32 + hi * 8;
  f.u[0] = *(const u32x4*)(p);
  f.u[1] = *(const u32x4*)(p + 16);
  return f.v;
}

__device__ __forceinline__ v16bf load_b(const unsigned* __restrict__ wf,
                                        int frag, int lane) {
  union { v16bf v; u32x4 u[2]; } f;
  const unsigned* q = wf + frag * 256 + lane * 8;
  f.u[0] = *(const u32x4*)(q);
  f.u[1] = *(const u32x4*)(q + 4);
  return f.v;
}

// C layout: VGPR i -> row m = i + 8*(lane>>4), col n = lane&15 (+ ntile*16)
__device__ __forceinline__ void store_relu(v8f acc, __bf16* dst, int mb, int ld,
                                           int n, float bias, int hi) {
#pragma unroll
  for (int i = 0; i < 8; ++i) {
    float v = fmaxf(acc[i] + bias, 0.f);
    dst[(mb * 16 + i + hi * 8) * ld + n] = (__bf16)v;
  }
}

#define WMMA_BF16(a, b, c) \
  __builtin_amdgcn_wmma_f32_16x16x32_bf16(false, (a), false, (b), (short)0, (c), false, false)

// ---------- pass 3: fused 3-layer MLP + epilogue ----------
// 128 threads = 4 waves cooperating on one 64-row tile; waves split the N dim,
// so each B fragment (from L2) feeds 4 WMMAs (M=64) -> 4x less L2 traffic/WMMA.
__global__ void __launch_bounds__(128)
k_mlp(const float* __restrict__ state, const float* __restrict__ action,
      const float* __restrict__ eps,
      const unsigned* __restrict__ w1f, const float* __restrict__ b1,
      const unsigned* __restrict__ w2f, const float* __restrict__ b2,
      const unsigned* __restrict__ w3f, const float* __restrict__ b3,
      const int* __restrict__ cnt, const int* __restrict__ rows,
      float* __restrict__ out) {
  extern __shared__ __bf16 smem[];            // [64 x HID_LD] H1 | [64 x HID_LD] X/H2
  const int lane = threadIdx.x & 31;
  const int w    = threadIdx.x >> 5;
  const int e    = blockIdx.x >> 9;           // / TILES(512)
  const int t    = blockIdx.x & (TILES - 1);
  const int cn   = cnt[e];
  const int m0   = t * TILE_M;
  if (m0 >= cn) return;                       // block-uniform exit
  __bf16* h1  = smem;
  __bf16* xh2 = smem + TILE_M * HID_LD;
  const int* erows = rows + e * B_TOTAL;
  const int hi = (lane >> 4) & 1;
  const int nl = lane & 15;

  // ---- stage X tile (64 x 96 bf16): [state | action | zero-pad]
  {
    int i = threadIdx.x & 63;
    int h = threadIdx.x >> 6;
    int rid = erows[m0 + i] & (B_TOTAL - 1);  // garbage rows clamped; stores masked later
    const float* srow = state + rid * SDIM;
    const float* arow = action + rid * ADIM;
    for (int c = h * 48; c < h * 48 + 48; ++c) {
      float v = (c < SDIM) ? srow[c] : ((c < SDIM + ADIM) ? arow[c - SDIM] : 0.f);
      xh2[i * IN_LD + c] = (__bf16)v;
    }
  }
  __syncthreads();

  // ---- layer 1: X(64x96) x W1(96x512) -> relu -> H1
  for (int nt = w; nt < 32; nt += 4) {
    v8f acc[4] = {};
#pragma unroll
    for (int kt = 0; kt < 3; ++kt) {
      v16bf b = load_b(w1f, (e * 32 + nt) * 3 + kt, lane);
#pragma unroll
      for (int mb = 0; mb < 4; ++mb) {
        v16bf a = load_a(xh2 + mb * 16 * IN_LD, IN_LD, kt, lane);
        acc[mb] = WMMA_BF16(a, b, acc[mb]);
      }
    }
    int n = nt * 16 + nl;
    float bias = b1[e * HID + n];
#pragma unroll
    for (int mb = 0; mb < 4; ++mb) store_relu(acc[mb], h1, mb, HID_LD, n, bias, hi);
  }
  __syncthreads();

  // ---- layer 2: H1(64x512) x W2(512x512) -> relu -> H2 (overwrites X region)
  for (int nt = w; nt < 32; nt += 4) {
    v8f acc[4] = {};
#pragma unroll 2
    for (int kt = 0; kt < 16; ++kt) {
      v16bf b = load_b(w2f, (e * 32 + nt) * 16 + kt, lane);
#pragma unroll
      for (int mb = 0; mb < 4; ++mb) {
        v16bf a = load_a(h1 + mb * 16 * HID_LD, HID_LD, kt, lane);
        acc[mb] = WMMA_BF16(a, b, acc[mb]);
      }
    }
    int n = nt * 16 + nl;
    float bias = b2[e * HID + n];
#pragma unroll
    for (int mb = 0; mb < 4; ++mb) store_relu(acc[mb], xh2, mb, HID_LD, n, bias, hi);
  }
  __syncthreads();

  // ---- layer 3: H2(64x512) x W3(512x144) -> O f32 (reuses H1 region: 64*144*4 B)
  float* O = (float*)h1;
  for (int nt = w; nt < 9; nt += 4) {
    v8f acc[4] = {};
#pragma unroll 2
    for (int kt = 0; kt < 16; ++kt) {
      v16bf b = load_b(w3f, (e * 9 + nt) * 16 + kt, lane);
#pragma unroll
      for (int mb = 0; mb < 4; ++mb) {
        v16bf a = load_a(xh2 + mb * 16 * HID_LD, HID_LD, kt, lane);
        acc[mb] = WMMA_BF16(a, b, acc[mb]);
      }
    }
    int n = nt * 16 + nl;
    float bias = (n < ODIM) ? b3[e * ODIM + n] : 0.f;
#pragma unroll
    for (int mb = 0; mb < 4; ++mb) {
#pragma unroll
      for (int i = 0; i < 8; ++i)
        O[(mb * 16 + i + hi * 8) * OPAD + n] = acc[mb][i] + bias;
    }
  }
  __syncthreads();

  // ---- epilogue: mu=O[:, :65], log_std=O[:, 65:130]; scatter to d_out
  float* out_ns = out;
  float* out_rw = out + B_TOTAL * SDIM;
  for (int tt = threadIdx.x; tt < TILE_M * 65; tt += 128) {
    int i = tt / 65;
    int d = tt - i * 65;
    int gr = m0 + i;
    if (gr >= cn) continue;                   // mask garbage rows (no WMMA below)
    int rid = erows[gr] & (B_TOTAL - 1);
    float mu = O[i * OPAD + d];
    float ls = O[i * OPAD + 65 + d];
    ls = fminf(2.0f, fmaxf(-20.0f, ls));
    float y = mu + __expf(ls) * eps[rid * 65 + d];
    if (d < SDIM) out_ns[rid * SDIM + d] = state[rid * SDIM + d] + y;
    else          out_rw[rid] = y;
  }
}

// ---------- host launcher ----------
extern "C" void kernel_launch(void* const* d_in, const int* in_sizes, int n_in,
                              void* d_out, int out_size, void* d_ws, size_t ws_size,
                              hipStream_t stream) {
  const float* state  = (const float*)d_in[0];
  const float* action = (const float*)d_in[1];
  const float* eps    = (const float*)d_in[2];
  const float* W1     = (const float*)d_in[3];
  const float* b1     = (const float*)d_in[4];
  const float* W2     = (const float*)d_in[5];
  const float* b2     = (const float*)d_in[6];
  const float* W3     = (const float*)d_in[7];
  const float* b3     = (const float*)d_in[8];
  const int*   idx    = (const int*)d_in[9];
  float* out = (float*)d_out;

  char* ws = (char*)d_ws;
  int* cnt  = (int*)(ws + OFF_CNT);
  int* rows = (int*)(ws + OFF_ROWS);
  unsigned* w1f = (unsigned*)(ws + OFF_W1F);
  unsigned* w2f = (unsigned*)(ws + OFF_W2F);
  unsigned* w3f = (unsigned*)(ws + OFF_W3F);

  hipLaunchKernelGGL(k_zero, dim3(1), dim3(32), 0, stream, cnt);
  hipLaunchKernelGGL(k_scatter, dim3(B_TOTAL / 256), dim3(256), 0, stream,
                     idx, cnt, rows);
  hipLaunchKernelGGL(k_convert, dim3((N_W1F + 255) / 256), dim3(256), 0, stream,
                     W1, w1f, 80, 512, 32, 3, N_W1F);
  hipLaunchKernelGGL(k_convert, dim3((N_W2F + 255) / 256), dim3(256), 0, stream,
                     W2, w2f, 512, 512, 32, 16, N_W2F);
  hipLaunchKernelGGL(k_convert, dim3((N_W3F + 255) / 256), dim3(256), 0, stream,
                     W3, w3f, 512, 130, 9, 16, N_W3F);
  hipLaunchKernelGGL(k_mlp, dim3(ENS * TILES), dim3(128), SMEM_BYTES, stream,
                     state, action, eps, w1f, b1, w2f, b2, w3f, b3, cnt, rows, out);
}
